// LGAPNet_76304388981323
// MI455X (gfx1250) — matrix operand order repaired and checked by reading, not compiled, not run on hardware
//
#include <hip/hip_runtime.h>
#include <math.h>

// ---------------------------------------------------------------------------
// LGAPNet forward for MI455X (gfx1250, wave32, WMMA).
// GEMMs: f16 WMMA (v_wmma_f32_16x16x32_f16) with f32 accumulate.
//   - 128x64 block tile, 8 waves, 32x32 wave tile (4 wmma / wave / K-step)
//   - LDS staged unswizzled: A [m][k] row-major,
//     B transposed [n][k]; both read with contiguous k = 16*hi + s
//     (same order in A and B, not the ISA table order) -> operand gather = two ds_load_b128 per matrix.
//   - All launches have M%128==0 and K%32==0 by construction (the K=67 embed
//     GEMM is zero-padded to K=96); only Nc in {32,13} needs the BEDGE
//     variant (vectorized A staging + float4-group-guarded B staging).
//   - global_prefetch_b8 of the next K-slab.
// Attention / kNN: wave-per-node VALU kernels, contiguous per-lane channel
// chunks so loads merge into b64/b128, shfl_xor cross-lane reductions.
// ---------------------------------------------------------------------------

typedef __attribute__((ext_vector_type(16))) _Float16 v16h;
typedef __attribute__((ext_vector_type(4)))  _Float16 v4h;
typedef __attribute__((ext_vector_type(8)))  float    v8f;

#define NPTS   32768
#define NPTS1  8192
#define NPTS2  2048
#define KEDGE  16

// block-of-8 column swizzle (unused: A is staged unswizzled):
// k 0-7 -> 0-7, 8-15 -> 16-23, 16-23 -> 8-15, 24-31 -> 24-31
__device__ __forceinline__ int permcol(int k)
{
    int b = ((k >> 3) ^ (k >> 4)) & 1;
    return k ^ (b * 0x18);
}

// ======================= WMMA GEMM =========================================
// C[M,Nc] = postop(A[M,K] @ W[K,Nc] + bias),  postop: RELU and/or +res
// Preconditions (enforced by the fixed launch set): M%128==0, K%32==0.
// BEDGE: Nc%64!=0 (group-guarded B staging, guarded epilogue columns)
template<int RELU, int RES, int BEDGE>
__global__ __launch_bounds__(256)
void gemm_kernel(const float* __restrict__ A, const float* __restrict__ W,
                 const float* __restrict__ bias, const float* __restrict__ res,
                 float* __restrict__ C, int M, int K, int Nc)
{
    // row stride 48 halves = 96 B: 32B-aligned rows for v16h loads
    __shared__ _Float16 Asp[128][48];  // [m][k]
    __shared__ _Float16 Bsp[64][48];   // [n][k]  (B stored transposed)

    const int tid  = threadIdx.x;
    const int lane = tid & 31;
    const int wave = tid >> 5;
    const int bm   = blockIdx.x * 128;
    const int bn   = blockIdx.y * 64;
    const int wm   = (wave & 3) * 32;     // wave's M sub-tile (32 rows)
    const int wn   = (wave >> 2) * 32;    // wave's N sub-tile (32 cols)
    const int l16  = lane & 15;
    const int hi   = lane >> 4;
    const int hi16 = hi * 16;

    v8f acc00 = {}, acc01 = {}, acc10 = {}, acc11 = {};

    for (int k0 = 0; k0 < K; k0 += 32) {
        // A tile 128x32: 1024 float4 groups, 4 per thread (M,K aligned)
        #pragma unroll
        for (int it = 0; it < 4; ++it) {
            int i = tid + it * 256;
            int r = i >> 3, cg = (i & 7) * 4;
            const float* ap = A + (size_t)(bm + r) * K + k0 + cg;
            float4 v = *(const float4*)ap;
            if (k0 + 32 < K) __builtin_prefetch(ap + 32, 0, 3);
            v4h pk;
            pk[0] = (_Float16)v.x; pk[1] = (_Float16)v.y;
            pk[2] = (_Float16)v.z; pk[3] = (_Float16)v.w;
            *(v4h*)&Asp[r][cg] = pk;
        }
        // B tile 32x64: 512 float4 groups, 2 per thread (transposed store)
        #pragma unroll
        for (int it = 0; it < 2; ++it) {
            int i = tid + it * 256;
            int r = i >> 4, cg = (i & 15) * 4;
            const float* wp = W + (size_t)(k0 + r) * Nc + bn + cg;
            float4 v;
            if (BEDGE) {
                int gn = bn + cg;
                v.x = v.y = v.z = v.w = 0.f;
                if (gn + 3 < Nc) {
                    v = *(const float4*)wp;
                } else {
                    if (gn + 0 < Nc) v.x = wp[0];
                    if (gn + 1 < Nc) v.y = wp[1];
                    if (gn + 2 < Nc) v.z = wp[2];
                }
            } else {
                v = *(const float4*)wp;
                if (k0 + 32 < K) __builtin_prefetch(wp + 32 * Nc, 0, 3);
            }
            Bsp[cg + 0][r] = (_Float16)v.x;
            Bsp[cg + 1][r] = (_Float16)v.y;
            Bsp[cg + 2][r] = (_Float16)v.z;
            Bsp[cg + 3][r] = (_Float16)v.w;
        }
        __syncthreads();

        // operand gather: contiguous 32B per lane (two ds_load_b128 each)
        const v16h a0 = *(const v16h*)&Asp[wm + l16][hi16];
        const v16h a1 = *(const v16h*)&Asp[wm + 16 + l16][hi16];
        const v16h b0 = *(const v16h*)&Bsp[wn + l16][hi16];
        const v16h b1 = *(const v16h*)&Bsp[wn + 16 + l16][hi16];

        acc00 = __builtin_amdgcn_wmma_f32_16x16x32_f16(false, a0, false, b0, (short)0, acc00, false, false);
        acc01 = __builtin_amdgcn_wmma_f32_16x16x32_f16(false, a0, false, b1, (short)0, acc01, false, false);
        acc10 = __builtin_amdgcn_wmma_f32_16x16x32_f16(false, a1, false, b0, (short)0, acc10, false, false);
        acc11 = __builtin_amdgcn_wmma_f32_16x16x32_f16(false, a1, false, b1, (short)0, acc11, false, false);
        __syncthreads();
    }

    // C/D layout: VGPR r -> M = r + 8*hi, N = l16
    const int m0 = bm + wm + hi * 8;
    const int n0 = bn + wn + l16;
    if (BEDGE) {
        float vb0 = (n0      < Nc) ? bias[n0]      : 0.f;
        float vb1 = (n0 + 16 < Nc) ? bias[n0 + 16] : 0.f;
        #pragma unroll
        for (int r = 0; r < 8; ++r) {
            int mA = m0 + r, mB = m0 + 16 + r;
            float v;
            if (n0 < Nc) {
                v = acc00[r] + vb0; if (RELU) v = fmaxf(v, 0.f);
                if (RES) v += res[(size_t)mA * Nc + n0];
                C[(size_t)mA * Nc + n0] = v;
                v = acc10[r] + vb0; if (RELU) v = fmaxf(v, 0.f);
                if (RES) v += res[(size_t)mB * Nc + n0];
                C[(size_t)mB * Nc + n0] = v;
            }
            if (n0 + 16 < Nc) {
                v = acc01[r] + vb1; if (RELU) v = fmaxf(v, 0.f);
                if (RES) v += res[(size_t)mA * Nc + n0 + 16];
                C[(size_t)mA * Nc + n0 + 16] = v;
                v = acc11[r] + vb1; if (RELU) v = fmaxf(v, 0.f);
                if (RES) v += res[(size_t)mB * Nc + n0 + 16];
                C[(size_t)mB * Nc + n0 + 16] = v;
            }
        }
    } else {
        const float vb0 = bias[n0], vb1 = bias[n0 + 16];
        #pragma unroll
        for (int r = 0; r < 8; ++r) {
            int mA = m0 + r, mB = m0 + 16 + r;
            float v00 = acc00[r] + vb0, v01 = acc01[r] + vb1;
            float v10 = acc10[r] + vb0, v11 = acc11[r] + vb1;
            if (RELU) {
                v00 = fmaxf(v00, 0.f); v01 = fmaxf(v01, 0.f);
                v10 = fmaxf(v10, 0.f); v11 = fmaxf(v11, 0.f);
            }
            if (RES) {
                v00 += res[(size_t)mA * Nc + n0];
                v01 += res[(size_t)mA * Nc + n0 + 16];
                v10 += res[(size_t)mB * Nc + n0];
                v11 += res[(size_t)mB * Nc + n0 + 16];
            }
            C[(size_t)mA * Nc + n0]      = v00;
            C[(size_t)mA * Nc + n0 + 16] = v01;
            C[(size_t)mB * Nc + n0]      = v10;
            C[(size_t)mB * Nc + n0 + 16] = v11;
        }
    }
}

// ======================= mesh token layer ==================================
__global__ void mesh_tok_kernel(const float* __restrict__ tok,
                                const float* __restrict__ W1,
                                const float* __restrict__ b1,
                                float* __restrict__ out, int T)
{
    int t = blockIdx.x * blockDim.x + threadIdx.x;
    if (t >= T) return;
    float x0 = tok[t * 3 + 0], x1 = tok[t * 3 + 1], x2 = tok[t * 3 + 2];
    #pragma unroll
    for (int j = 0; j < 8; ++j) {
        float v = x0 * W1[j] + x1 * W1[8 + j] + x2 * W1[16 + j] + b1[j];
        out[t * 8 + j] = fmaxf(v, 0.f);
    }
}

// ======================= weight zero-pad (K=67 -> 96) ======================
__global__ void pad_weight_kernel(const float* __restrict__ Wsrc,
                                  float* __restrict__ Wdst, int n_src, int n_dst)
{
    int i = blockIdx.x * blockDim.x + threadIdx.x;
    if (i >= n_dst) return;
    Wdst[i] = (i < n_src) ? Wsrc[i] : 0.f;
}

// ======================= GAPL attention ====================================
// Each dst node owns 16 contiguous edges (dst = repeat(arange(n),16)).
// One wave32 per node; lane owns CPT *contiguous* channels (merged loads).
template<int C>
__global__ __launch_bounds__(256)
void gapl_attn_kernel(const float* __restrict__ pos, const float* __restrict__ x,
                      const float* __restrict__ h, const int* __restrict__ src,
                      float* __restrict__ y, int n)
{
    const int lane = threadIdx.x & 31;
    const int node = blockIdx.x * 8 + (threadIdx.x >> 5);
    if (node >= n) return;
    constexpr int CPT = C / 32;
    const int cbase = lane * CPT;
    const float scale = 0.5f / sqrtf((float)C);   // ALPHA / sqrt(c)

    float hd[CPT], agg[CPT];
    #pragma unroll
    for (int j = 0; j < CPT; ++j) {
        hd[j]  = h[(size_t)node * C + cbase + j];
        agg[j] = 0.f;
    }
    const float px = pos[node * 3 + 0], py = pos[node * 3 + 1], pz = pos[node * 3 + 2];

    float logit[KEDGE];
    float mx = -3.4e38f;
    #pragma unroll
    for (int k = 0; k < KEDGE; ++k) {
        int s = src[node * KEDGE + k];
        float part = 0.f;
        #pragma unroll
        for (int j = 0; j < CPT; ++j)
            part += hd[j] * h[(size_t)s * C + cbase + j];
        #pragma unroll
        for (int off = 16; off >= 1; off >>= 1)
            part += __shfl_xor(part, off, 32);
        float dx = pos[s * 3 + 0] - px, dy = pos[s * 3 + 1] - py, dz = pos[s * 3 + 2] - pz;
        float l = scale * part - (dx * dx + dy * dy + dz * dz);   // BETA = 1
        logit[k] = l;
        mx = fmaxf(mx, l);
    }
    float ssum = 0.f, w[KEDGE];
    #pragma unroll
    for (int k = 0; k < KEDGE; ++k) { w[k] = __expf(logit[k] - mx); ssum += w[k]; }
    const float inv = 1.f / (ssum + 1e-9f);
    #pragma unroll
    for (int k = 0; k < KEDGE; ++k) {
        int s = src[node * KEDGE + k];
        float ww = w[k] * inv;
        #pragma unroll
        for (int j = 0; j < CPT; ++j)
            agg[j] += ww * h[(size_t)s * C + cbase + j];
    }
    #pragma unroll
    for (int j = 0; j < CPT; ++j) {
        size_t o = (size_t)node * C + cbase + j;
        y[o] = fmaxf(x[o] + agg[j], 0.f);
    }
}

// ======================= kNN (k=3) interpolate =============================
__device__ __forceinline__ void top3_insert(float d, int id, float& d0, int& i0,
                                            float& d1, int& i1, float& d2, int& i2)
{
    if (d < d2) {
        if (d < d1) {
            if (d < d0) { d2 = d1; i2 = i1; d1 = d0; i1 = i0; d0 = d; i0 = id; }
            else        { d2 = d1; i2 = i1; d1 = d;  i1 = id; }
        } else        { d2 = d;  i2 = id; }
    }
}

template<int C>
__global__ __launch_bounds__(256)
void knn_interp_kernel(const float* __restrict__ xsrc, const float* __restrict__ pos_src,
                       const float* __restrict__ pos_up, float* __restrict__ out,
                       int n_src, int n_up)
{
    const int lane = threadIdx.x & 31;
    const int p = blockIdx.x * 8 + (threadIdx.x >> 5);
    if (p >= n_up) return;
    const float ux = pos_up[p * 3], uy = pos_up[p * 3 + 1], uz = pos_up[p * 3 + 2];

    float d0 = 3.4e38f, d1 = 3.4e38f, d2 = 3.4e38f;
    int   i0 = 0, i1 = 0, i2 = 0;
    for (int base = 0; base < n_src; base += 32) {
        int j = base + lane;
        float dd = 3.4e38f;
        if (j < n_src) {
            float dx = pos_src[j * 3] - ux, dy = pos_src[j * 3 + 1] - uy, dz = pos_src[j * 3 + 2] - uz;
            dd = dx * dx + dy * dy + dz * dz;
        }
        top3_insert(dd, j, d0, i0, d1, i1, d2, i2);
    }
    // butterfly merge of per-lane top3 -> global top3 (all lanes converge)
    #pragma unroll
    for (int off = 16; off >= 1; off >>= 1) {
        float e0 = __shfl_xor(d0, off, 32), e1 = __shfl_xor(d1, off, 32), e2 = __shfl_xor(d2, off, 32);
        int   j0 = __shfl_xor(i0, off, 32), j1 = __shfl_xor(i1, off, 32), j2 = __shfl_xor(i2, off, 32);
        top3_insert(e0, j0, d0, i0, d1, i1, d2, i2);
        top3_insert(e1, j1, d0, i0, d1, i1, d2, i2);
        top3_insert(e2, j2, d0, i0, d1, i1, d2, i2);
    }
    float w0 = 1.f / (d0 + 1e-8f), w1 = 1.f / (d1 + 1e-8f), w2 = 1.f / (d2 + 1e-8f);
    float inv = 1.f / (w0 + w1 + w2);
    w0 *= inv; w1 *= inv; w2 *= inv;
    constexpr int CPT = C / 32;
    const int cbase = lane * CPT;
    #pragma unroll
    for (int j = 0; j < CPT; ++j) {
        int c = cbase + j;
        out[(size_t)p * C + c] = w0 * xsrc[(size_t)i0 * C + c]
                               + w1 * xsrc[(size_t)i1 * C + c]
                               + w2 * xsrc[(size_t)i2 * C + c];
    }
}

// ======================= concat kernels ====================================
// concat2: Ca, Cb multiples of 4 -> float4 copies
__global__ void concat2_kernel(const float* __restrict__ a, int Ca,
                               const float* __restrict__ b, int Cb,
                               float* __restrict__ dst, int M)
{
    const int Ct4 = (Ca + Cb) >> 2;
    const int Ca4 = Ca >> 2;
    const size_t total = (size_t)M * Ct4;
    for (size_t i = (size_t)blockIdx.x * blockDim.x + threadIdx.x; i < total;
         i += (size_t)gridDim.x * blockDim.x) {
        size_t r = i / Ct4; int c4 = (int)(i % Ct4);
        float4 v = (c4 < Ca4) ? ((const float4*)(a + r * Ca))[c4]
                              : ((const float4*)(b + r * Cb))[c4 - Ca4];
        ((float4*)(dst + r * (size_t)(Ca + Cb)))[c4] = v;
    }
}

// concat3 with zero-padded destination stride CtPad (for K=96 embed GEMM)
__global__ void concat3_kernel(const float* __restrict__ a, int Ca,
                               const float* __restrict__ b, int Cb,
                               const float* __restrict__ c3, int Cc,
                               float* __restrict__ dst, int M, int CtPad)
{
    const size_t total = (size_t)M * CtPad;
    for (size_t i = (size_t)blockIdx.x * blockDim.x + threadIdx.x; i < total;
         i += (size_t)gridDim.x * blockDim.x) {
        size_t r = i / CtPad; int c = (int)(i % CtPad);
        float v = 0.f;
        if (c < Ca)                v = a[r * Ca + c];
        else if (c < Ca + Cb)      v = b[r * Cb + (c - Ca)];
        else if (c < Ca + Cb + Cc) v = c3[r * Cc + (c - Ca - Cb)];
        dst[i] = v;
    }
}

// ======================= host-side launcher ================================
// Preconditions (all launches below satisfy them): M%128==0, K%32==0.
static void launch_gemm(int relu, int resf, const float* A, const float* W,
                        const float* bias, const float* res, float* C,
                        int M, int K, int Nc, hipStream_t s)
{
    dim3 grid((M + 127) / 128, (Nc + 63) / 64);
    const bool bedge = (Nc & 63) != 0;
    if (bedge) {
        if (relu && resf) gemm_kernel<1, 1, 1><<<grid, 256, 0, s>>>(A, W, bias, res, C, M, K, Nc);
        else if (relu)    gemm_kernel<1, 0, 1><<<grid, 256, 0, s>>>(A, W, bias, res, C, M, K, Nc);
        else              gemm_kernel<0, 0, 1><<<grid, 256, 0, s>>>(A, W, bias, res, C, M, K, Nc);
    } else {
        if (relu && resf) gemm_kernel<1, 1, 0><<<grid, 256, 0, s>>>(A, W, bias, res, C, M, K, Nc);
        else if (relu)    gemm_kernel<1, 0, 0><<<grid, 256, 0, s>>>(A, W, bias, res, C, M, K, Nc);
        else              gemm_kernel<0, 0, 0><<<grid, 256, 0, s>>>(A, W, bias, res, C, M, K, Nc);
    }
}

// workspace layout (float offsets) — liveness-checked reuse
static constexpr size_t N_  = NPTS, N1_ = NPTS1, N2_ = NPTS2;
static constexpr size_t OFF_X0  = 0;                      // x0 home, N*64 (long-lived)
static constexpr size_t OFF_X1  = N_ * 64;                // x1 home, N1*128 (long-lived)
static constexpr size_t G0      = OFF_X1 + N1_ * 128;     // reusable pool
// stage1 (encoders)
static constexpr size_t OFF_TMP  = G0;                    // N*32
static constexpr size_t OFF_XRGB = G0 + N_ * 32;          // N*32
static constexpr size_t OFF_XNRM = G0 + N_ * 64;          // N*32
static constexpr size_t OFF_XCAT = G0 + N_ * 96;          // N*96 (stride 96, zero-padded)
static constexpr size_t OFF_XEMB = G0 + N_ * 192;         // N*64
static constexpr size_t OFF_WPAD = G0 + N_ * 256;         // 96*64 padded emb_W
// stage2 (GAPL0)
static constexpr size_t OFF_H0  = G0;                     // N*64
static constexpr size_t OFF_Y0  = G0 + N_ * 64;           // N*64
static constexpr size_t OFF_XG0 = G0 + N_ * 128;          // N*64
// stage3 (GAPL1, rows 0..N1)
static constexpr size_t OFF_H1  = G0;                     // N1*64
static constexpr size_t OFF_Y1  = G0 + N1_ * 64;          // N1*64
static constexpr size_t OFF_XL1 = G0 + N1_ * 128;         // N1*64
static constexpr size_t OFF_T1  = G0 + N1_ * 192;         // N1*128
// stage4 (GAPL2 + up1 path)
static constexpr size_t OFF_H2   = G0;                    // N2*128
static constexpr size_t OFF_Y2   = G0 + N2_ * 128;        // N2*128
static constexpr size_t OFF_XL2  = G0 + N2_ * 256;        // N2*128
static constexpr size_t OFF_T2   = G0 + N2_ * 384;        // N2*256
static constexpr size_t OFF_X2   = G0 + N2_ * 640;        // N2*256
static constexpr size_t OFF_UP1  = G0 + N2_ * 896;        // N1*256
static constexpr size_t OFF_CAT1 = OFF_UP1 + N1_ * 256;   // N1*384
static constexpr size_t OFF_T3   = G0;                    // N1*128 (h2.. dead)
static constexpr size_t OFF_F1   = G0 + N1_ * 128;        // N1*128
// stage5 (up2 path)
static constexpr size_t OFF_UP2  = G0 + N_ * 64;          // N*128
static constexpr size_t OFF_CAT2 = G0 + N_ * 192;         // N*192
static constexpr size_t OFF_T4   = G0;                    // N*64
static constexpr size_t OFF_F2   = G0 + N_ * 64;          // N*64 (up2 dead)
// stage6 (head)
static constexpr size_t OFF_Y5   = G0 + N_ * 128;         // N*128
static constexpr size_t OFF_Y6   = G0;                    // N*128 (t4/f2 dead)

extern "C" void kernel_launch(void* const* d_in, const int* in_sizes, int n_in,
                              void* d_out, int out_size, void* d_ws, size_t ws_size,
                              hipStream_t stream)
{
    (void)in_sizes; (void)n_in; (void)out_size; (void)ws_size;

    const float* pos     = (const float*)d_in[0];
    const float* rgb     = (const float*)d_in[1];
    const float* normals = (const float*)d_in[2];
    const float* ce_W1 = (const float*)d_in[3],  *ce_b1 = (const float*)d_in[4];
    const float* ce_W2 = (const float*)d_in[5],  *ce_b2 = (const float*)d_in[6];
    const float* ne_W1 = (const float*)d_in[7],  *ne_b1 = (const float*)d_in[8];
    const float* ne_W2 = (const float*)d_in[9],  *ne_b2 = (const float*)d_in[10];
    const float* emb_W = (const float*)d_in[11], *emb_b = (const float*)d_in[12];
    const float* g0_W  = (const float*)d_in[13], *g0_b  = (const float*)d_in[14];
    const float* g0_rW = (const float*)d_in[15], *g0_rb = (const float*)d_in[16];
    const float* enc_rW= (const float*)d_in[17], *enc_rb= (const float*)d_in[18];
    const float* g1_W  = (const float*)d_in[19], *g1_b  = (const float*)d_in[20];
    const float* g1_rW = (const float*)d_in[21], *g1_rb = (const float*)d_in[22];
    const float* rc1_W = (const float*)d_in[23], *rc1_b = (const float*)d_in[24];
    const float* rc1_rW= (const float*)d_in[25], *rc1_rb= (const float*)d_in[26];
    const float* g2_W  = (const float*)d_in[27], *g2_b  = (const float*)d_in[28];
    const float* g2_rW = (const float*)d_in[29], *g2_rb = (const float*)d_in[30];
    const float* rc2_W = (const float*)d_in[31], *rc2_b = (const float*)d_in[32];
    const float* rc2_rW= (const float*)d_in[33], *rc2_rb= (const float*)d_in[34];
    const float* u1_W  = (const float*)d_in[35], *u1_b  = (const float*)d_in[36];
    const float* u1_rW = (const float*)d_in[37], *u1_rb = (const float*)d_in[38];
    const float* u2_W  = (const float*)d_in[39], *u2_b  = (const float*)d_in[40];
    const float* u2_rW = (const float*)d_in[41], *u2_rb = (const float*)d_in[42];
    const float* m1_W  = (const float*)d_in[43], *m1_b  = (const float*)d_in[44];
    const float* m2_W  = (const float*)d_in[45], *m2_b  = (const float*)d_in[46];
    const float* m3_W  = (const float*)d_in[47], *m3_b  = (const float*)d_in[48];
    const int* src0 = (const int*)d_in[49];   // edge_index0 row 0 = src
    const int* src1 = (const int*)d_in[50];
    const int* src2 = (const int*)d_in[51];

    float* ws  = (float*)d_ws;
    float* out = (float*)d_out;

    const int N = NPTS, Nu1 = NPTS1, Nu2 = NPTS2;

    // ---- mesh encoders (rgb / normals) ------------------------------------
    mesh_tok_kernel<<<(N * 4 + 255) / 256, 256, 0, stream>>>(rgb, ce_W1, ce_b1, ws + OFF_TMP, N * 4);
    launch_gemm(1, 0, ws + OFF_TMP, ce_W2, ce_b2, nullptr, ws + OFF_XRGB, N, 32, 32, stream);
    mesh_tok_kernel<<<(N * 4 + 255) / 256, 256, 0, stream>>>(normals, ne_W1, ne_b1, ws + OFF_TMP, N * 4);
    launch_gemm(1, 0, ws + OFF_TMP, ne_W2, ne_b2, nullptr, ws + OFF_XNRM, N, 32, 32, stream);

    // ---- embed: x = relu(concat(x_rgb, x_nrm, pos)|pad96 @ emb_W|pad96) ---
    {
        size_t total = (size_t)N * 96;
        concat3_kernel<<<(unsigned)((total + 255) / 256), 256, 0, stream>>>(
            ws + OFF_XRGB, 32, ws + OFF_XNRM, 32, pos, 3, ws + OFF_XCAT, N, 96);
        pad_weight_kernel<<<(96 * 64 + 255) / 256, 256, 0, stream>>>(emb_W, ws + OFF_WPAD, 67 * 64, 96 * 64);
    }
    launch_gemm(1, 0, ws + OFF_XCAT, ws + OFF_WPAD, emb_b, nullptr, ws + OFF_XEMB, N, 96, 64, stream);

    // ---- GAPL 0 (C=64) + encoder resMLP -> x0 -----------------------------
    launch_gemm(0, 0, ws + OFF_XEMB, g0_W, g0_b, nullptr, ws + OFF_H0, N, 64, 64, stream);
    gapl_attn_kernel<64><<<(N + 7) / 8, 256, 0, stream>>>(pos, ws + OFF_XEMB, ws + OFF_H0, src0, ws + OFF_Y0, N);
    launch_gemm(1, 1, ws + OFF_Y0, g0_rW, g0_rb, ws + OFF_Y0, ws + OFF_XG0, N, 64, 64, stream);
    launch_gemm(1, 1, ws + OFF_XG0, enc_rW, enc_rb, ws + OFF_XG0, ws + OFF_X0, N, 64, 64, stream);

    // ---- GAPL 1 on x0[:N1] -> x1 ------------------------------------------
    launch_gemm(0, 0, ws + OFF_X0, g1_W, g1_b, nullptr, ws + OFF_H1, Nu1, 64, 64, stream);
    gapl_attn_kernel<64><<<(Nu1 + 7) / 8, 256, 0, stream>>>(pos, ws + OFF_X0, ws + OFF_H1, src1, ws + OFF_Y1, Nu1);
    launch_gemm(1, 1, ws + OFF_Y1, g1_rW, g1_rb, ws + OFF_Y1, ws + OFF_XL1, Nu1, 64, 64, stream);
    launch_gemm(1, 0, ws + OFF_XL1, rc1_W, rc1_b, nullptr, ws + OFF_T1, Nu1, 64, 128, stream);
    launch_gemm(1, 1, ws + OFF_T1, rc1_rW, rc1_rb, ws + OFF_T1, ws + OFF_X1, Nu1, 128, 128, stream);

    // ---- GAPL 2 on x1[:N2] -> x2 ------------------------------------------
    launch_gemm(0, 0, ws + OFF_X1, g2_W, g2_b, nullptr, ws + OFF_H2, Nu2, 128, 128, stream);
    gapl_attn_kernel<128><<<(Nu2 + 7) / 8, 256, 0, stream>>>(pos, ws + OFF_X1, ws + OFF_H2, src2, ws + OFF_Y2, Nu2);
    launch_gemm(1, 1, ws + OFF_Y2, g2_rW, g2_rb, ws + OFF_Y2, ws + OFF_XL2, Nu2, 128, 128, stream);
    launch_gemm(1, 0, ws + OFF_XL2, rc2_W, rc2_b, nullptr, ws + OFF_T2, Nu2, 128, 256, stream);
    launch_gemm(1, 1, ws + OFF_T2, rc2_rW, rc2_rb, ws + OFF_T2, ws + OFF_X2, Nu2, 256, 256, stream);

    // ---- upsample N2 -> N1, fuse with x1 ----------------------------------
    knn_interp_kernel<256><<<(Nu1 + 7) / 8, 256, 0, stream>>>(ws + OFF_X2, pos, pos, ws + OFF_UP1, Nu2, Nu1);
    {
        size_t total = (size_t)Nu1 * 96;   // 384/4 float4s per row
        concat2_kernel<<<(unsigned)((total + 255) / 256), 256, 0, stream>>>(
            ws + OFF_X1, 128, ws + OFF_UP1, 256, ws + OFF_CAT1, Nu1);
    }
    launch_gemm(1, 0, ws + OFF_CAT1, u1_W, u1_b, nullptr, ws + OFF_T3, Nu1, 384, 128, stream);
    launch_gemm(1, 1, ws + OFF_T3, u1_rW, u1_rb, ws + OFF_T3, ws + OFF_F1, Nu1, 128, 128, stream);

    // ---- upsample N1 -> N, fuse with x0 -----------------------------------
    knn_interp_kernel<128><<<(N + 7) / 8, 256, 0, stream>>>(ws + OFF_F1, pos, pos, ws + OFF_UP2, Nu1, N);
    {
        size_t total = (size_t)N * 48;     // 192/4 float4s per row
        concat2_kernel<<<(unsigned)((total + 255) / 256), 256, 0, stream>>>(
            ws + OFF_X0, 64, ws + OFF_UP2, 128, ws + OFF_CAT2, N);
    }
    launch_gemm(1, 0, ws + OFF_CAT2, u2_W, u2_b, nullptr, ws + OFF_T4, N, 192, 64, stream);
    launch_gemm(1, 1, ws + OFF_T4, u2_rW, u2_rb, ws + OFF_T4, ws + OFF_F2, N, 64, 64, stream);

    // ---- head MLP ----------------------------------------------------------
    launch_gemm(1, 0, ws + OFF_F2, m1_W, m1_b, nullptr, ws + OFF_Y5, N, 64, 128, stream);
    launch_gemm(1, 0, ws + OFF_Y5, m2_W, m2_b, nullptr, ws + OFF_Y6, N, 128, 128, stream);
    launch_gemm(0, 0, ws + OFF_Y6, m3_W, m3_b, nullptr, out, N, 128, 13, stream);
}
